// FusionLSTMPerformer_18640158064967
// MI455X (gfx1250) — matrix operand order repaired and checked by reading, not compiled, not run on hardware
//
#include <hip/hip_runtime.h>
#include <math.h>

// ---------------------------------------------------------------------------
// Model constants (from reference)
// ---------------------------------------------------------------------------
#define BB    8
#define TT    512
#define DIN   256
#define HH    512
#define DD    1024          // d_model
#define NHEAD 8
#define DH    128
#define MF    256           // nb_features
#define FFD   4096
#define ROWS  (BB*TT)       // 4096 flat (b,t) rows

typedef __bf16 bf16;
typedef __attribute__((ext_vector_type(16))) __bf16 v16bf;
typedef __attribute__((ext_vector_type(8)))  float  v8f;
typedef int int4v __attribute__((vector_size(16)));   // matches builtin param type

// Optional gfx1250 async global->LDS copy (guarded; falls back to sync copy)
#if defined(__has_builtin)
#  if __has_builtin(__builtin_amdgcn_global_load_async_to_lds_b128)
#    define HAVE_ASYNC_LDS 1
#  endif
#  if __has_builtin(__builtin_amdgcn_s_wait_asynccnt)
#    define HAVE_WAIT_ASYNC 1
#  endif
#endif
#ifndef HAVE_ASYNC_LDS
#  define HAVE_ASYNC_LDS 0
#endif
#ifndef HAVE_WAIT_ASYNC
#  define HAVE_WAIT_ASYNC 0
#endif

union ABfrag { v16bf v; uint4 q[2]; };

// A-matrix fragment (16x32 bf16). Per ISA: lane row = lane%16, khalf = lane/16,
// element e -> k = (e/8)*16 + khalf*8 + e%8  => two contiguous 16B runs.
__device__ __forceinline__ v16bf load_a_frag(const bf16* ap, int khalf) {
  ABfrag f;
  f.q[0] = *(const uint4*)(ap + khalf * 8);
  f.q[1] = *(const uint4*)(ap + 16 + khalf * 8);
  return f.v;
}
// B-matrix fragment (32x16 bf16), W stored [N][K]. Same k order as A:
// element e -> k = (e/8)*16 + khalf*8 + e%8  => two contiguous 16B runs.
__device__ __forceinline__ v16bf load_b_frag(const bf16* bp, int khalf) {
  ABfrag f;
  f.q[0] = *(const uint4*)(bp + khalf * 8);
  f.q[1] = *(const uint4*)(bp + 16 + khalf * 8);
  return f.v;
}

__device__ __forceinline__ float blk_sum(float v, float* sm, int tid, int n) {
  sm[tid] = v; __syncthreads();
  for (int o = n >> 1; o > 0; o >>= 1) {
    if (tid < o) sm[tid] += sm[tid + o];
    __syncthreads();
  }
  float r = sm[0]; __syncthreads();
  return r;
}
__device__ __forceinline__ float blk_max(float v, float* sm, int tid, int n) {
  sm[tid] = v; __syncthreads();
  for (int o = n >> 1; o > 0; o >>= 1) {
    if (tid < o) sm[tid] = fmaxf(sm[tid], sm[tid + o]);
    __syncthreads();
  }
  float r = sm[0]; __syncthreads();
  return r;
}
__device__ __forceinline__ float sigmf(float x) { return 1.f / (1.f + __expf(-x)); }

// ---------------------------------------------------------------------------
// Elementwise f32 -> bf16 convert (optional scale)
// ---------------------------------------------------------------------------
__global__ void cvt_k(const float* __restrict__ in, bf16* __restrict__ out,
                      float scale, long n) {
  long i = (long)blockIdx.x * blockDim.x + threadIdx.x;
  long stride = (long)gridDim.x * blockDim.x;
  for (; i < n; i += stride) out[i] = (bf16)(in[i] * scale);
}

__global__ void addvec_k(const float* a, const float* b, float* o, int n) {
  int i = blockIdx.x * blockDim.x + threadIdx.x;
  if (i < n) o[i] = a[i] + b[i];
}

// ---------------------------------------------------------------------------
// Batched transpose + convert: out_bf16[z][C][R] = in_f32[zoff][R][C]
// ---------------------------------------------------------------------------
__global__ void trans_k(const float* __restrict__ in, bf16* __restrict__ out,
                        int ldi, int ldo, long is0, long is1, int bdiv, long os) {
  int z = blockIdx.z;
  int zq = (bdiv > 1) ? z / bdiv : z;
  int zr = (bdiv > 1) ? z % bdiv : 0;
  const float* ip = in + (size_t)zq * is0 + (size_t)zr * is1;
  bf16* op = out + (size_t)z * os;
  __shared__ float tile[32][33];
  int c0 = blockIdx.x * 32, r0 = blockIdx.y * 32;
  for (int i = 0; i < 4; ++i) {
    int r = threadIdx.y + i * 8;
    tile[r][threadIdx.x] = ip[(size_t)(r0 + r) * ldi + c0 + threadIdx.x];
  }
  __syncthreads();
  for (int i = 0; i < 4; ++i) {
    int r = threadIdx.y + i * 8;
    op[(size_t)(c0 + r) * ldo + r0 + threadIdx.x] = (bf16)tile[threadIdx.x][r];
  }
}

// ---------------------------------------------------------------------------
// Generic batched bf16 WMMA GEMM:  C = act(A[M][K] @ Bt[N][K]^T + bias) + res
//   - 256 threads (8 waves), 64x128 tile per workgroup, K stepped by 32
//   - each wave: 1 A-frag (LDS) + 4 B-tiles (global/L2) -> 4 WMMA per slab
//   - A tile staged via async global->LDS copy when available
//   - M multiple of 64, N multiple of 128, K multiple of 32 (all shapes here ok)
// ---------------------------------------------------------------------------
struct GemmP {
  const bf16* A; const bf16* Bt; const float* bias; const float* res;
  float* C; bf16* Cb;
  long aS0, aS1, bS0, cS0, cS1, biasS0;
  int lda, ldb, ldc, K, bdiv;
};

template <int ACT>  // 0=none, 1=gelu(exact), 2=relu
__global__ __launch_bounds__(256) void gemm_k(GemmP p) {
  int z  = blockIdx.z;
  int zq = (p.bdiv > 1) ? z / p.bdiv : z;
  int zr = (p.bdiv > 1) ? z % p.bdiv : 0;
  const bf16* A = p.A + (size_t)zq * p.aS0 + (size_t)zr * p.aS1;
  const bf16* B = p.Bt + (size_t)zq * p.bS0;
  size_t cOff   = (size_t)zq * p.cS0 + (size_t)zr * p.cS1;

  int tid = threadIdx.x, lane = tid & 31, wave = tid >> 5;
  int mt = wave & 3, nq = wave >> 2;          // 4 M-tiles x 2 N-quadrants
  int row = lane & 15, khalf = lane >> 4;
  int by = blockIdx.y * 64, bx = blockIdx.x * 128;

  __shared__ bf16 aS[64][40];   // 32-col K slab, padded row stride (80B, 16B mult)
  v8f acc[4] = {};

  for (int k0 = 0; k0 < p.K; k0 += 32) {
    {  // stage A tile: 64 rows x 32 bf16, one b128 per thread
      int r = tid >> 2, g = tid & 3;
      const bf16* src = A + (size_t)(by + r) * p.lda + k0 + g * 8;
      bf16* dst = &aS[r][g * 8];
#if HAVE_ASYNC_LDS
      __builtin_amdgcn_global_load_async_to_lds_b128(
          (__attribute__((address_space(1))) int4v*)src,
          (__attribute__((address_space(3))) int4v*)dst, 0, 0);
#if HAVE_WAIT_ASYNC
      __builtin_amdgcn_s_wait_asynccnt(0);
#else
      asm volatile("s_wait_asynccnt 0x0" ::: "memory");
#endif
#else
      *(uint4*)dst = *(const uint4*)src;
#endif
    }
    __syncthreads();
    v16bf a = load_a_frag(&aS[mt * 16 + row][0], khalf);
    int n0 = bx + nq * 64 + row;
    const bf16* bp = B + (size_t)n0 * p.ldb + k0;
    __builtin_prefetch(bp + 32, 0, 1);  // next K slab of B (global_prefetch)
#pragma unroll
    for (int j = 0; j < 4; ++j) {
      v16bf b = load_b_frag(bp + (size_t)(j * 16) * p.ldb, khalf);
      acc[j] = __builtin_amdgcn_wmma_f32_16x16x32_bf16(false, a, false, b,
                                                       (short)0, acc[j], false, false);
    }
    __syncthreads();
  }

  int n0 = bx + nq * 64 + row;
  for (int j = 0; j < 4; ++j) {
    int n = n0 + j * 16;
    float bv = p.bias ? p.bias[(size_t)zq * p.biasS0 + n] : 0.f;
    for (int r8 = 0; r8 < 8; ++r8) {
      int m = by + mt * 16 + khalf * 8 + r8;
      float v = acc[j][r8] + bv;
      if (ACT == 1) v = 0.5f * v * (1.f + erff(v * 0.70710678118654752f));
      if (ACT == 2) v = fmaxf(v, 0.f);
      size_t ci = cOff + (size_t)m * p.ldc + n;
      if (p.res) v += p.res[ci];
      if (p.C)  p.C[ci]  = v;
      if (p.Cb) p.Cb[ci] = (bf16)v;
    }
  }
}

// ---------------------------------------------------------------------------
// LSTM recurrence: one 32-wave workgroup per direction. Per step:
//   z[8x2048] = h[8x512] @ Whh^T (WMMA, Whh stored [2048][512] bf16)
//             + precomputed (x@Wih^T + bih + bhh)
//   gates -> c,h update; h kept as bf16 A-matrix in LDS.
// NOTE: the K loop must stay a dynamic loop (#pragma unroll 2). Full unroll
// lets LICM hoist all Whh fragments across the 512-step loop -> massive
// scratch spills (seen in round-1 disasm). Dynamic loop keeps B fragments
// streaming from L2 (global_load_b128 -> v_wmma), which is the intended path.
// ---------------------------------------------------------------------------
__global__ __launch_bounds__(1024) void lstm_rec_k(const float* __restrict__ zbuf,
                                                   const bf16* __restrict__ Whh,
                                                   float* __restrict__ out) {
  int dir = blockIdx.x;
  const float* zb = zbuf + (size_t)dir * ROWS * (4 * HH);
  const bf16*  W  = Whh  + (size_t)dir * (4 * HH) * HH;

  __shared__ bf16 hS[16][528];   // padded rows (1056B, 16B mult); rows 8..15 = 0
  __shared__ bf16 zS[8][2048];

  int tid = threadIdx.x, lane = tid & 31, wave = tid >> 5;
  int row = lane & 15, khalf = lane >> 4;

  for (int i = tid; i < 16 * 528; i += 1024) ((bf16*)hS)[i] = (bf16)0.f;
  float c[4] = {0.f, 0.f, 0.f, 0.f};
  int bu = tid * 4;  // 4 (b,u) pairs per thread; 8*512 total
  __syncthreads();

  for (int step = 0; step < TT; ++step) {
    int t = dir ? (TT - 1 - step) : step;

    v8f acc[4] = {};
#pragma unroll 2
    for (int k0 = 0; k0 < HH; k0 += 32) {
      v16bf a = load_a_frag(&hS[row][k0], khalf);
      __builtin_prefetch(W + (size_t)((wave * 4) * 16 + row) * HH + k0 + 32, 0, 1);
#pragma unroll
      for (int j = 0; j < 4; ++j) {
        int n = (wave * 4 + j) * 16 + row;
        v16bf b = load_b_frag(W + (size_t)n * HH + k0, khalf);
        acc[j] = __builtin_amdgcn_wmma_f32_16x16x32_bf16(false, a, false, b,
                                                         (short)0, acc[j], false, false);
      }
    }
    if (khalf == 0) {  // only rows 0..7 are real batch rows
      for (int j = 0; j < 4; ++j) {
        int n = (wave * 4 + j) * 16 + row;
        for (int r8 = 0; r8 < 8; ++r8) {
          int m = r8;
          float zv = acc[j][r8] + zb[(size_t)(m * TT + t) * (4 * HH) + n];
          zS[m][n] = (bf16)zv;
        }
      }
    }
    __syncthreads();

    // gate phase: thread owns 4 fixed (b,u) cells (c stays in registers)
    float hv[4];
    for (int i2 = 0; i2 < 4; ++i2) {
      int idx = bu + i2;
      int b = idx >> 9, u = idx & 511;
      float zi = (float)zS[b][u];
      float zf = (float)zS[b][512 + u];
      float zg = (float)zS[b][1024 + u];
      float zo = (float)zS[b][1536 + u];
      float cv = sigmf(zf) * c[i2] + sigmf(zi) * tanhf(zg);
      c[i2] = cv;
      float h = sigmf(zo) * tanhf(cv);
      hv[i2] = h;
      out[(size_t)(b * TT + t) * DD + dir * HH + u] = h;
    }
    for (int i2 = 0; i2 < 4; ++i2) {
      int idx = bu + i2;
      hS[idx >> 9][idx & 511] = (bf16)hv[i2];
    }
    __syncthreads();
  }
}

// ---------------------------------------------------------------------------
// LayerNorm over D=1024, writes bf16 (feeds WMMA GEMMs)
// ---------------------------------------------------------------------------
__global__ __launch_bounds__(256) void ln_k(const float* __restrict__ x,
                                            const float* __restrict__ g,
                                            const float* __restrict__ b,
                                            bf16* __restrict__ y) {
  int r = blockIdx.x, tid = threadIdx.x;
  const float* xr = x + (size_t)r * DD;
  __shared__ float sm[256];
  float s = 0.f;
  for (int i = tid; i < DD; i += 256) s += xr[i];
  float mu = blk_sum(s, sm, tid, 256) * (1.f / DD);
  float s2 = 0.f;
  for (int i = tid; i < DD; i += 256) { float d = xr[i] - mu; s2 += d * d; }
  float var = blk_sum(s2, sm, tid, 256) * (1.f / DD);
  float rstd = rsqrtf(var + 1e-5f);
  for (int i = tid; i < DD; i += 256)
    y[(size_t)r * DD + i] = (bf16)((xr[i] - mu) * rstd * g[i] + b[i]);
}

// diag[b,h,t] = sum_d q[b,t,h*128+d]^2 * (dn^2 * 0.5),  dn^2 = 128^-0.5
__global__ __launch_bounds__(128) void diag_k(const bf16* __restrict__ qk,
                                              float* __restrict__ diag) {
  int idx = blockIdx.x;             // bh*512 + t
  int bh = idx >> 9, t = idx & 511;
  int b = bh >> 3, h = bh & 7;
  const bf16* r = qk + (size_t)(b * TT + t) * DD + h * DH;
  float v = (float)r[threadIdx.x];
  __shared__ float sm[128];
  float s = blk_sum(v * v, sm, threadIdx.x, 128);
  if (threadIdx.x == 0) diag[idx] = s * 0.044194173824159216f;
}

// query phi: per-row max over M features
__global__ __launch_bounds__(256) void qphi_k(const float* __restrict__ dd,
                                              const float* __restrict__ diag,
                                              bf16* __restrict__ qphi) {
  int r = blockIdx.x, m = threadIdx.x;
  __shared__ float sm[256];
  float v = dd[(size_t)r * MF + m];
  float mx = blk_max(v, sm, m, 256);
  float phi = (__expf(v - diag[r] - mx) + 1e-4f) * 0.0625f;  // * M^-0.5
  qphi[(size_t)r * MF + m] = (bf16)phi;
}

__global__ __launch_bounds__(256) void kmax_k(const float* __restrict__ dd,
                                              float* __restrict__ kmax) {
  int bh = blockIdx.x, tid = threadIdx.x;
  const float* p = dd + (size_t)bh * TT * MF;
  float m = -1e30f;
  for (int i = tid; i < TT * MF; i += 256) m = fmaxf(m, p[i]);
  __shared__ float sm[256];
  m = blk_max(m, sm, tid, 256);
  if (tid == 0) kmax[bh] = m;
}

// key phi: global (T,M) max per (b,h); writes transposed kphiT[bh][m][t] (bf16)
__global__ __launch_bounds__(256) void kphi_k(const float* __restrict__ dd,
                                              const float* __restrict__ diag,
                                              const float* __restrict__ kmax,
                                              bf16* __restrict__ kphiT) {
  int r = blockIdx.x, m = threadIdx.x;
  int bh = r >> 9, t = r & 511;
  float v = dd[(size_t)r * MF + m];
  float phi = (__expf(v - diag[r] - kmax[bh]) + 1e-4f) * 0.0625f;
  kphiT[((size_t)bh * MF + m) * TT + t] = (bf16)phi;
}

// ksum[bh][m] = sum_t kphiT[bh][m][t]
__global__ __launch_bounds__(256) void ksum_k(const bf16* __restrict__ kphiT,
                                              float* __restrict__ ksum) {
  int id = blockIdx.x, tid = threadIdx.x;     // id = bh*256 + m
  const bf16* p = kphiT + (size_t)id * TT;
  float v = (float)p[tid] + (float)p[tid + 256];
  __shared__ float sm[256];
  float s = blk_sum(v, sm, tid, 256);
  if (tid == 0) ksum[id] = s;
}

// dinv = 1/(qphi . ksum); scale o and scatter to flat [B*T][D] bf16
__global__ __launch_bounds__(256) void dinv_k(const bf16* __restrict__ qphi,
                                              const float* __restrict__ ksum,
                                              const float* __restrict__ of,
                                              bf16* __restrict__ oflat) {
  int r = blockIdx.x, tid = threadIdx.x;      // r = bh*512 + t
  int bh = r >> 9, t = r & 511;
  float v = (float)qphi[(size_t)r * MF + tid] * ksum[bh * MF + tid];
  __shared__ float sm[256];
  float dot = blk_sum(v, sm, tid, 256);
  float dinv = 1.f / dot;
  if (tid < DH) {
    float ov = of[(size_t)r * DH + tid] * dinv;
    int b = bh >> 3, h = bh & 7;
    oflat[(size_t)(b * TT + t) * DD + h * DH + tid] = (bf16)ov;
  }
}

__global__ __launch_bounds__(256) void pool_k(const float* __restrict__ h,
                                              float* __restrict__ pooled) {
  int b = blockIdx.x;
  for (int d = threadIdx.x; d < DD; d += 256) {
    float s = 0.f;
    for (int t = 0; t < TT; ++t) s += h[(size_t)(b * TT + t) * DD + d];
    pooled[b * DD + d] = s * (1.f / TT);
  }
}

__global__ __launch_bounds__(512) void cls_k(const float* __restrict__ pooled,
                                             const float* __restrict__ W1,
                                             const float* __restrict__ b1,
                                             const float* __restrict__ W2,
                                             const float* __restrict__ b2,
                                             float* __restrict__ out) {
  int b = blockIdx.x, u = threadIdx.x;
  __shared__ float hid[HH];
  const float* pb = pooled + b * DD;
  float s = b1[u];
  for (int d = 0; d < DD; ++d) s += pb[d] * W1[d * HH + u];
  hid[u] = fmaxf(s, 0.f);
  __syncthreads();
  if (u < 2) {
    float o = b2[u];
    for (int j = 0; j < HH; ++j) o += hid[j] * W2[j * 2 + u];
    out[b * 2 + u] = o;
  }
}

// ---------------------------------------------------------------------------
// Host orchestration
// ---------------------------------------------------------------------------
static inline void launch_gemm(hipStream_t s, const GemmP& p, int M, int N,
                               int batch, int act) {
  dim3 g(N / 128, M / 64, batch), blk(256);
  if (act == 1)      hipLaunchKernelGGL((gemm_k<1>), g, blk, 0, s, p);
  else if (act == 2) hipLaunchKernelGGL((gemm_k<2>), g, blk, 0, s, p);
  else               hipLaunchKernelGGL((gemm_k<0>), g, blk, 0, s, p);
}

extern "C" void kernel_launch(void* const* d_in, const int* in_sizes, int n_in,
                              void* d_out, int out_size, void* d_ws, size_t ws_size,
                              hipStream_t stream) {
  (void)in_sizes; (void)n_in; (void)out_size; (void)ws_size;
  const float* x      = (const float*)d_in[0];
  const float* l0Wih  = (const float*)d_in[1];
  const float* l0Whh  = (const float*)d_in[2];
  const float* l0bih  = (const float*)d_in[3];
  const float* l0bhh  = (const float*)d_in[4];
  const float* l1Wih  = (const float*)d_in[5];
  const float* l1Whh  = (const float*)d_in[6];
  const float* l1bih  = (const float*)d_in[7];
  const float* l1bhh  = (const float*)d_in[8];
  const float* proj   = (const float*)d_in[9];
  const float* ln1g   = (const float*)d_in[10];
  const float* ln1b   = (const float*)d_in[11];
  const float* Wq     = (const float*)d_in[12];
  const float* Wk     = (const float*)d_in[13];
  const float* Wv     = (const float*)d_in[14];
  const float* Wo     = (const float*)d_in[15];
  const float* bq     = (const float*)d_in[16];
  const float* bk     = (const float*)d_in[17];
  const float* bv     = (const float*)d_in[18];
  const float* bo     = (const float*)d_in[19];
  const float* ln2g   = (const float*)d_in[20];
  const float* ln2b   = (const float*)d_in[21];
  const float* Wf1    = (const float*)d_in[22];
  const float* bf1    = (const float*)d_in[23];
  const float* Wf2    = (const float*)d_in[24];
  const float* bf2    = (const float*)d_in[25];
  const float* cW1    = (const float*)d_in[26];
  const float* cb1    = (const float*)d_in[27];
  const float* cW2    = (const float*)d_in[28];
  const float* cb2    = (const float*)d_in[29];
  float* logits = (float*)d_out;

  // bump allocator over workspace (256B aligned)
  char* wp = (char*)d_ws;
  auto alloc = [&](size_t bytes) -> void* {
    void* r = (void*)wp;
    wp += (bytes + 255) & ~(size_t)255;
    return r;
  };
  bf16*  x_bf    = (bf16*)alloc((size_t)ROWS * DIN * 2);
  bf16*  wih0_bf = (bf16*)alloc((size_t)2 * 4 * HH * DIN * 2);
  bf16*  whh0_bf = (bf16*)alloc((size_t)2 * 4 * HH * HH * 2);
  bf16*  wih1_bf = (bf16*)alloc((size_t)2 * 4 * HH * DD * 2);
  bf16*  whh1_bf = (bf16*)alloc((size_t)2 * 4 * HH * HH * 2);
  bf16*  projS   = (bf16*)alloc((size_t)MF * DH * 2);
  bf16*  WqT     = (bf16*)alloc((size_t)2 * DD * DD * 2);
  bf16*  WkT     = (bf16*)alloc((size_t)2 * DD * DD * 2);
  bf16*  WvT     = (bf16*)alloc((size_t)2 * DD * DD * 2);
  bf16*  WoT     = (bf16*)alloc((size_t)2 * DD * DD * 2);
  bf16*  Wf1T    = (bf16*)alloc((size_t)2 * FFD * DD * 2);
  bf16*  Wf2T    = (bf16*)alloc((size_t)2 * DD * FFD * 2);
  float* bsum0   = (float*)alloc((size_t)2 * 4 * HH * 4);
  float* bsum1   = (float*)alloc((size_t)2 * 4 * HH * 4);
  float* zbuf    = (float*)alloc((size_t)2 * ROWS * 4 * HH * 4);
  float* h0      = (float*)alloc((size_t)ROWS * DD * 4);
  bf16*  h0_bf   = (bf16*)alloc((size_t)ROWS * DD * 2);
  float* h       = (float*)alloc((size_t)ROWS * DD * 4);
  bf16*  y_bf    = (bf16*)alloc((size_t)ROWS * DD * 2);
  bf16*  q_bf    = (bf16*)alloc((size_t)ROWS * DD * 2);
  bf16*  k_bf    = (bf16*)alloc((size_t)ROWS * DD * 2);
  float* v_f     = (float*)alloc((size_t)ROWS * DD * 4);
  bf16*  vT_bf   = (bf16*)alloc((size_t)BB * NHEAD * DH * TT * 2);
  float* diag    = (float*)alloc((size_t)BB * NHEAD * TT * 4);
  float* dd      = (float*)alloc((size_t)BB * NHEAD * TT * MF * 4);
  bf16*  qphi    = (bf16*)alloc((size_t)BB * NHEAD * TT * MF * 2);
  bf16*  kphiT   = (bf16*)alloc((size_t)BB * NHEAD * MF * TT * 2);
  float* kmax    = (float*)alloc((size_t)BB * NHEAD * 4);
  float* ksum    = (float*)alloc((size_t)BB * NHEAD * MF * 4);
  bf16*  ctxT    = (bf16*)alloc((size_t)BB * NHEAD * DH * MF * 2);
  float* o_f     = (float*)alloc((size_t)BB * NHEAD * TT * DH * 4);
  bf16*  oflat   = (bf16*)alloc((size_t)ROWS * DD * 2);
  bf16*  ff_bf   = (bf16*)alloc((size_t)ROWS * FFD * 2);
  float* pooled  = (float*)alloc((size_t)BB * DD * 4);

  const float DN = 0.29730177875068026f;  // 128^-0.25 (folded into proj)

  // ---- weight conversion / transposition ----
  auto cvt = [&](const float* in, bf16* out, float sc, long n) {
    int blocks = (int)((n + 255) / 256); if (blocks > 4096) blocks = 4096;
    hipLaunchKernelGGL(cvt_k, dim3(blocks), dim3(256), 0, stream, in, out, sc, n);
  };
  cvt(x, x_bf, 1.f, (long)ROWS * DIN);
  cvt(l0Wih, wih0_bf, 1.f, (long)2 * 4 * HH * DIN);
  cvt(l0Whh, whh0_bf, 1.f, (long)2 * 4 * HH * HH);
  cvt(l1Wih, wih1_bf, 1.f, (long)2 * 4 * HH * DD);
  cvt(l1Whh, whh1_bf, 1.f, (long)2 * 4 * HH * HH);
  cvt(proj, projS, DN, (long)MF * DH);

  auto transW = [&](const float* in, bf16* out, int R, int C, long batchStride) {
    hipLaunchKernelGGL(trans_k, dim3(C / 32, R / 32, 2), dim3(32, 8), 0, stream,
                       in, out, C, R, batchStride, 0L, 1, batchStride);
  };
  transW(Wq, WqT, DD, DD, (long)DD * DD);
  transW(Wk, WkT, DD, DD, (long)DD * DD);
  transW(Wv, WvT, DD, DD, (long)DD * DD);
  transW(Wo, WoT, DD, DD, (long)DD * DD);
  transW(Wf1, Wf1T, DD, FFD, (long)DD * FFD);
  transW(Wf2, Wf2T, FFD, DD, (long)FFD * DD);

  hipLaunchKernelGGL(addvec_k, dim3(16), dim3(256), 0, stream, l0bih, l0bhh, bsum0, 2 * 4 * HH);
  hipLaunchKernelGGL(addvec_k, dim3(16), dim3(256), 0, stream, l1bih, l1bhh, bsum1, 2 * 4 * HH);

  // ---- BiLSTM layer 0 ----
  {
    GemmP p = {};
    p.A = x_bf; p.Bt = wih0_bf; p.bias = bsum0;
    p.C = zbuf; p.bS0 = (long)4 * HH * DIN; p.cS0 = (long)ROWS * 4 * HH;
    p.biasS0 = 4 * HH; p.lda = DIN; p.ldb = DIN; p.ldc = 4 * HH; p.K = DIN; p.bdiv = 1;
    launch_gemm(stream, p, ROWS, 4 * HH, 2, 0);
  }
  hipLaunchKernelGGL(lstm_rec_k, dim3(2), dim3(1024), 0, stream, zbuf, whh0_bf, h0);
  cvt(h0, h0_bf, 1.f, (long)ROWS * DD);

  // ---- BiLSTM layer 1 ----
  {
    GemmP p = {};
    p.A = h0_bf; p.Bt = wih1_bf; p.bias = bsum1;
    p.C = zbuf; p.bS0 = (long)4 * HH * DD; p.cS0 = (long)ROWS * 4 * HH;
    p.biasS0 = 4 * HH; p.lda = DD; p.ldb = DD; p.ldc = 4 * HH; p.K = DD; p.bdiv = 1;
    launch_gemm(stream, p, ROWS, 4 * HH, 2, 0);
  }
  hipLaunchKernelGGL(lstm_rec_k, dim3(2), dim3(1024), 0, stream, zbuf, whh1_bf, h);

  // ---- Performer blocks ----
  const int BH = BB * NHEAD;
  for (int i = 0; i < 2; ++i) {
    hipLaunchKernelGGL(ln_k, dim3(ROWS), dim3(256), 0, stream,
                       h, ln1g + i * DD, ln1b + i * DD, y_bf);
    // q, k, v projections
    {
      GemmP p = {};
      p.A = y_bf; p.Bt = WqT + (size_t)i * DD * DD; p.bias = bq + i * DD;
      p.Cb = q_bf; p.lda = DD; p.ldb = DD; p.ldc = DD; p.K = DD; p.bdiv = 1;
      launch_gemm(stream, p, ROWS, DD, 1, 0);
      p.Bt = WkT + (size_t)i * DD * DD; p.bias = bk + i * DD; p.Cb = k_bf;
      launch_gemm(stream, p, ROWS, DD, 1, 0);
      p.Bt = WvT + (size_t)i * DD * DD; p.bias = bv + i * DD; p.Cb = nullptr; p.C = v_f;
      launch_gemm(stream, p, ROWS, DD, 1, 0);
    }
    // vT[bh][d][t]
    hipLaunchKernelGGL(trans_k, dim3(DH / 32, TT / 32, BH), dim3(32, 8), 0, stream,
                       v_f, vT_bf, DD, TT, (long)TT * DD, (long)DH, NHEAD, (long)DH * TT);
    // query features
    hipLaunchKernelGGL(diag_k, dim3(BH * TT), dim3(128), 0, stream, q_bf, diag);
    {
      GemmP p = {};
      p.A = q_bf; p.aS0 = (long)TT * DD; p.aS1 = DH; p.Bt = projS;
      p.C = dd; p.cS0 = (long)NHEAD * TT * MF; p.cS1 = (long)TT * MF;
      p.lda = DD; p.ldb = DH; p.ldc = MF; p.K = DH; p.bdiv = NHEAD;
      launch_gemm(stream, p, TT, MF, BH, 0);
    }
    hipLaunchKernelGGL(qphi_k, dim3(BH * TT), dim3(256), 0, stream, dd, diag, qphi);
    // key features
    hipLaunchKernelGGL(diag_k, dim3(BH * TT), dim3(128), 0, stream, k_bf, diag);
    {
      GemmP p = {};
      p.A = k_bf; p.aS0 = (long)TT * DD; p.aS1 = DH; p.Bt = projS;
      p.C = dd; p.cS0 = (long)NHEAD * TT * MF; p.cS1 = (long)TT * MF;
      p.lda = DD; p.ldb = DH; p.ldc = MF; p.K = DH; p.bdiv = NHEAD;
      launch_gemm(stream, p, TT, MF, BH, 0);
    }
    hipLaunchKernelGGL(kmax_k, dim3(BH), dim3(256), 0, stream, dd, kmax);
    hipLaunchKernelGGL(kphi_k, dim3(BH * TT), dim3(256), 0, stream, dd, diag, kmax, kphiT);
    hipLaunchKernelGGL(ksum_k, dim3(BH * MF), dim3(256), 0, stream, kphiT, ksum);
    // ctxT[d][m] = sum_n vT[d][n] * kphi[n][m]
    {
      GemmP p = {};
      p.A = vT_bf; p.aS0 = (long)DH * TT; p.Bt = kphiT; p.bS0 = (long)MF * TT;
      p.Cb = ctxT; p.cS0 = (long)DH * MF;
      p.lda = TT; p.ldb = TT; p.ldc = MF; p.K = TT; p.bdiv = 1;
      launch_gemm(stream, p, DH, MF, BH, 0);
    }
    // o[t][d] = qphi[t][:] . ctxT[d][:]
    {
      GemmP p = {};
      p.A = qphi; p.aS0 = (long)TT * MF; p.Bt = ctxT; p.bS0 = (long)DH * MF;
      p.C = o_f; p.cS0 = (long)TT * DH;
      p.lda = MF; p.ldb = MF; p.ldc = DH; p.K = MF; p.bdiv = 1;
      launch_gemm(stream, p, TT, DH, BH, 0);
    }
    hipLaunchKernelGGL(dinv_k, dim3(BH * TT), dim3(256), 0, stream, qphi, ksum, o_f, oflat);
    // h = h + o @ Wo + bo   (in-place residual)
    {
      GemmP p = {};
      p.A = oflat; p.Bt = WoT + (size_t)i * DD * DD; p.bias = bo + i * DD;
      p.res = h; p.C = h; p.lda = DD; p.ldb = DD; p.ldc = DD; p.K = DD; p.bdiv = 1;
      launch_gemm(stream, p, ROWS, DD, 1, 0);
    }
    // FF: h = h + W2 @ gelu(W1 @ ln2(h))
    hipLaunchKernelGGL(ln_k, dim3(ROWS), dim3(256), 0, stream,
                       h, ln2g + i * DD, ln2b + i * DD, y_bf);
    {
      GemmP p = {};
      p.A = y_bf; p.Bt = Wf1T + (size_t)i * FFD * DD; p.bias = bf1 + i * FFD;
      p.Cb = ff_bf; p.lda = DD; p.ldb = DD; p.ldc = FFD; p.K = DD; p.bdiv = 1;
      launch_gemm(stream, p, ROWS, FFD, 1, 1);  // exact GELU
    }
    {
      GemmP p = {};
      p.A = ff_bf; p.Bt = Wf2T + (size_t)i * DD * FFD; p.bias = bf2 + i * DD;
      p.res = h; p.C = h; p.lda = FFD; p.ldb = FFD; p.ldc = DD; p.K = FFD; p.bdiv = 1;
      launch_gemm(stream, p, ROWS, DD, 1, 0);
    }
  }

  // ---- pooling + classifier ----
  hipLaunchKernelGGL(pool_k, dim3(BB), dim3(256), 0, stream, h, pooled);
  hipLaunchKernelGGL(cls_k, dim3(BB), dim3(512), 0, stream,
                     pooled, cW1, cb1, cW2, cb2, logits);
}
